// PatchReaderComplex_89653147336988
// MI455X (gfx1250) — matrix operand order repaired and checked by reading, not compiled, not run on hardware
//
#include <hip/hip_runtime.h>

// ---------------------------------------------------------------------------
// GCN pipeline for MI455X (gfx1250, wave32):
//   conv1 (scatter over 64 feats) -> WMMA bf16 GEMM 64->128 + leaky
//   graphnorm1 (fused affine)     -> WMMA bf16 GEMM 128->64 (W-first trick)
//   conv2 (scatter over 64 feats) -> leaky + graphnorm2
//   segment-mean -> tiny MLP readout (50 rows)
// ---------------------------------------------------------------------------

typedef __bf16 v16bf __attribute__((ext_vector_type(16)));
typedef float  v8f   __attribute__((ext_vector_type(8)));

#define LEAKY(x) ((x) > 0.f ? (x) : 0.01f * (x))
#define G_EPS 1e-5f

// ---------------- degree / normalization ----------------------------------
__global__ void deg_kernel(const int* __restrict__ src, const int* __restrict__ dst,
                           float* __restrict__ outd, float* __restrict__ ind, int E) {
  int e = blockIdx.x * blockDim.x + threadIdx.x;
  if (e < E) {
    atomicAdd(&outd[src[e]], 1.f);
    atomicAdd(&ind[dst[e]], 1.f);
  }
}

__global__ void invnorm_kernel(float* __restrict__ outd, float* __restrict__ ind, int N) {
  int n = blockIdx.x * blockDim.x + threadIdx.x;
  if (n < N) {
    float o = outd[n]; outd[n] = rsqrtf(o > 0.f ? o : 1.f);
    float i = ind[n];  ind[n]  = rsqrtf(i > 0.f ? i : 1.f);
  }
}

// ---------------- weight convert + transpose (fp32 -> bf16) ----------------
// W is [K, Nout] row-major; WT becomes [Nout, K] so B-fragments load contiguously.
__global__ void cvtT_kernel(const float* __restrict__ W, __bf16* __restrict__ WT,
                            int K, int Nout) {
  int idx = blockIdx.x * blockDim.x + threadIdx.x;
  if (idx < K * Nout) {
    int k = idx / Nout, n = idx % Nout;
    WT[(size_t)n * K + k] = (__bf16)W[idx];
  }
}

// ---------------- edge scatter (64 features, float4 per thread) ------------
__global__ void scatter_kernel(const float* __restrict__ feat, const int* __restrict__ src,
                               const int* __restrict__ dst, const float* __restrict__ ew,
                               const float* __restrict__ rowscale, float* __restrict__ agg,
                               int E) {
  int tid = blockIdx.x * blockDim.x + threadIdx.x;
  int e = tid >> 4;
  if (e >= E) return;
  int q = tid & 15;
  int s = src[e], d = dst[e];
  float c = ew[e];
  if (rowscale) c *= rowscale[s];
  const float4 v = *(const float4*)(feat + (size_t)s * 64 + q * 4);
  float* o = agg + (size_t)d * 64 + q * 4;
  atomicAdd(o + 0, v.x * c);
  atomicAdd(o + 1, v.y * c);
  atomicAdd(o + 2, v.z * c);
  atomicAdd(o + 3, v.w * c);
}

// ---------------- GEMM1: h1 = leaky( (agg * in_n) @ W1 )  [N,64]@[64,128] ---
// one wave per 16x16 tile; blockDim = (32, 8) -> 8 column tiles per block.
__global__ void gemm1_kernel(const float* __restrict__ agg, const float* __restrict__ in_n,
                             const __bf16* __restrict__ W1T, float* __restrict__ h1) {
  const int lane = threadIdx.x;
  const int hi = lane >> 4, lo = lane & 15;
  const int row0 = blockIdx.x * 16;
  const int col0 = threadIdx.y * 16;
  const int m = row0 + lo;
  const float scale = in_n[m];
  const float* arow = agg + (size_t)m * 64;
  v8f c = {};
#pragma unroll
  for (int kb = 0; kb < 64; kb += 32) {
    // A fragment (16x32 bf16): elem e -> K = kb + 16*hi + e
    float4 p0 = *(const float4*)(arow + kb + hi * 16);
    float4 p1 = *(const float4*)(arow + kb + hi * 16 + 4);
    float4 p2 = *(const float4*)(arow + kb + hi * 16 + 8);
    float4 p3 = *(const float4*)(arow + kb + hi * 16 + 12);
    float av[16] = {p0.x, p0.y, p0.z, p0.w, p1.x, p1.y, p1.z, p1.w,
                    p2.x, p2.y, p2.z, p2.w, p3.x, p3.y, p3.z, p3.w};
    v16bf a;
#pragma unroll
    for (int e = 0; e < 16; ++e) a[e] = (__bf16)(av[e] * scale);
    // B fragment (32x16 bf16): elem e -> K = kb + 16*hi + e, contiguous in W1T
    v16bf b = *(const v16bf*)(W1T + (size_t)(col0 + lo) * 64 + kb + hi * 16);
    c = __builtin_amdgcn_wmma_f32_16x16x32_bf16(false, a, false, b, (short)0, c,
                                                false, false);
  }
#pragma unroll
  for (int r = 0; r < 8; ++r) {
    int row = row0 + hi * 8 + r;
    float v = c[r];
    h1[(size_t)row * 128 + col0 + lo] = LEAKY(v);
  }
}

// ---------------- column stats (sum, sum-of-squares) -----------------------
template <int F>
__global__ void colstats_kernel(const float* __restrict__ x, float* __restrict__ sum,
                                float* __restrict__ sq, int N) {
  constexpr int RPB = 256 / F;
  const int c = threadIdx.x % F;
  const int rg = threadIdx.x / F;
  float s = 0.f, q = 0.f;
  for (int r = blockIdx.x * RPB + rg; r < N; r += gridDim.x * RPB) {
    float v = x[(size_t)r * F + c];
    s += v; q += v * v;
  }
  __shared__ float ss[256], qq[256];
  ss[threadIdx.x] = s; qq[threadIdx.x] = q;
  __syncthreads();
  if (rg == 0) {
#pragma unroll
    for (int g = 1; g < RPB; ++g) { s += ss[g * F + c]; q += qq[g * F + c]; }
    atomicAdd(&sum[c], s);
    atomicAdd(&sq[c], q);
  }
}

// graphnorm -> per-column affine y = a*x + b
__global__ void coeffs_kernel(const float* __restrict__ sum, const float* __restrict__ sq,
                              const float* __restrict__ gamma, const float* __restrict__ beta,
                              const float* __restrict__ alpha, float* __restrict__ a,
                              float* __restrict__ b, int F, float invN) {
  int c = blockIdx.x * blockDim.x + threadIdx.x;
  if (c < F) {
    float m = sum[c] * invN;
    float ex2 = sq[c] * invN;
    float al = alpha[c];
    float var = ex2 - 2.f * al * m * m + al * al * m * m;
    float aa = gamma[c] * rsqrtf(var + G_EPS);
    a[c] = aa;
    b[c] = beta[c] - aa * al * m;
  }
}

// ---------------- GEMM2: t2 = ((a1*h1+b1) * out_n) @ W2  [N,128]@[128,64] ---
__global__ void gemm2_kernel(const float* __restrict__ h1, const float* __restrict__ out_n,
                             const float* __restrict__ a1, const float* __restrict__ b1,
                             const __bf16* __restrict__ W2T, float* __restrict__ t2) {
  const int lane = threadIdx.x;
  const int hi = lane >> 4, lo = lane & 15;
  const int row0 = blockIdx.x * 16;
  const int col0 = threadIdx.y * 16;
  const int m = row0 + lo;
  const float scale = out_n[m];
  const float* hrow = h1 + (size_t)m * 128;
  v8f c = {};
#pragma unroll
  for (int kb = 0; kb < 128; kb += 32) {
    v16bf a;
#pragma unroll
    for (int e = 0; e < 8; ++e) {
      int k = kb + hi * 16 + e;
      a[e] = (__bf16)((a1[k] * hrow[k] + b1[k]) * scale);
    }
#pragma unroll
    for (int e = 0; e < 8; ++e) {
      int k = kb + hi * 16 + 8 + e;
      a[8 + e] = (__bf16)((a1[k] * hrow[k] + b1[k]) * scale);
    }
    v16bf b = *(const v16bf*)(W2T + (size_t)(col0 + lo) * 128 + kb + hi * 16);
    c = __builtin_amdgcn_wmma_f32_16x16x32_bf16(false, a, false, b, (short)0, c,
                                                false, false);
  }
#pragma unroll
  for (int r = 0; r < 8; ++r) {
    int row = row0 + hi * 8 + r;
    t2[(size_t)row * 64 + col0 + lo] = c[r];
  }
}

// conv2 epilogue: h2 = leaky(agg2 * in_n)  (in-place on agg2)
__global__ void finish2_kernel(float* __restrict__ agg2, const float* __restrict__ in_n,
                               int total) {
  int i = blockIdx.x * blockDim.x + threadIdx.x;
  if (i < total) {
    int n = i >> 6;
    float v = agg2[i] * in_n[n];
    agg2[i] = LEAKY(v);
  }
}

// segment mean accumulation (graphnorm2 affine fused in)
__global__ void segment_kernel(const float* __restrict__ h2, const int* __restrict__ gid,
                               const float* __restrict__ a2, const float* __restrict__ b2,
                               float* __restrict__ gsum, float* __restrict__ gcnt, int N) {
  int tid = blockIdx.x * blockDim.x + threadIdx.x;
  int n = tid >> 4;
  if (n >= N) return;
  int q = tid & 15;
  int g = gid[n];
  const float* hr = h2 + (size_t)n * 64;
  float* o = gsum + (size_t)g * 64 + q * 4;
#pragma unroll
  for (int j = 0; j < 4; ++j) {
    int f = q * 4 + j;
    atomicAdd(o + j, a2[f] * hr[f] + b2[f]);
  }
  if (q == 0) atomicAdd(&gcnt[g], 1.f);
}

// ---------------- readout MLP: one 64-thread block per graph ---------------
__global__ void readout_kernel(const float* __restrict__ gsum, const float* __restrict__ gcnt,
                               const float* __restrict__ L1, const float* __restrict__ L2,
                               const float* __restrict__ L3, const float* __restrict__ Wc,
                               float* __restrict__ out) {
  __shared__ float x0[64], x1[32], x2[16], x3[8];
  __shared__ float mrs[2];
  const int r = blockIdx.x, t = threadIdx.x;
  float cnt = fmaxf(gcnt[r], 1.f);
  x0[t] = gsum[(size_t)r * 64 + t] / cnt;
  __syncthreads();
  if (t < 32) {
    float acc = 0.f;
    for (int k = 0; k < 64; ++k) acc += x0[k] * L1[k * 32 + t];
    x1[t] = LEAKY(acc);
  }
  __syncthreads();
  if (t == 0) {
    float m = 0.f; for (int k = 0; k < 32; ++k) m += x1[k]; m /= 32.f;
    float v = 0.f; for (int k = 0; k < 32; ++k) { float d = x1[k] - m; v += d * d; } v /= 32.f;
    mrs[0] = m; mrs[1] = rsqrtf(v + G_EPS);
  }
  __syncthreads();
  if (t < 32) x1[t] = (x1[t] - mrs[0]) * mrs[1];
  __syncthreads();
  if (t < 16) {
    float acc = 0.f;
    for (int k = 0; k < 32; ++k) acc += x1[k] * L2[k * 16 + t];
    x2[t] = LEAKY(acc);
  }
  __syncthreads();
  if (t == 0) {
    float m = 0.f; for (int k = 0; k < 16; ++k) m += x2[k]; m /= 16.f;
    float v = 0.f; for (int k = 0; k < 16; ++k) { float d = x2[k] - m; v += d * d; } v /= 16.f;
    mrs[0] = m; mrs[1] = rsqrtf(v + G_EPS);
  }
  __syncthreads();
  if (t < 16) x2[t] = (x2[t] - mrs[0]) * mrs[1];
  __syncthreads();
  if (t < 8) {
    float acc = 0.f;
    for (int k = 0; k < 16; ++k) acc += x2[k] * L3[k * 8 + t];
    x3[t] = LEAKY(acc);
  }
  __syncthreads();
  if (t == 0) {
    float m = 0.f; for (int k = 0; k < 8; ++k) m += x3[k]; m /= 8.f;
    float v = 0.f; for (int k = 0; k < 8; ++k) { float d = x3[k] - m; v += d * d; } v /= 8.f;
    mrs[0] = m; mrs[1] = rsqrtf(v + G_EPS);
  }
  __syncthreads();
  if (t < 8) x3[t] = (x3[t] - mrs[0]) * mrs[1];
  __syncthreads();
  if (t < 16) {
    float acc = 0.f;
    for (int k = 0; k < 8; ++k) acc += x3[k] * Wc[k * 16 + t];
    out[(size_t)r * 16 + t] = acc;
  }
}

// ---------------------------------------------------------------------------
extern "C" void kernel_launch(void* const* d_in, const int* in_sizes, int n_in,
                              void* d_out, int out_size, void* d_ws, size_t ws_size,
                              hipStream_t stream) {
  const float* node_feats = (const float*)d_in[0];
  const int*   src        = (const int*)d_in[1];
  const int*   dst        = (const int*)d_in[2];
  const float* ew         = (const float*)d_in[3];
  const int*   gid        = (const int*)d_in[4];
  const float* W1         = (const float*)d_in[5];
  const float* gn1_gamma  = (const float*)d_in[6];
  const float* gn1_beta   = (const float*)d_in[7];
  const float* gn1_alpha  = (const float*)d_in[8];
  const float* W2         = (const float*)d_in[9];
  const float* gn2_gamma  = (const float*)d_in[10];
  const float* gn2_beta   = (const float*)d_in[11];
  const float* gn2_alpha  = (const float*)d_in[12];
  const float* L1         = (const float*)d_in[13];
  const float* L2         = (const float*)d_in[14];
  const float* L3         = (const float*)d_in[15];
  const float* Wc         = (const float*)d_in[16];
  float* out = (float*)d_out;

  const int N = in_sizes[0] / 64;  // 50000 (divisible by 16)
  const int E = in_sizes[1];       // 800000
  const int B = 50;

  // 256B-aligned sub-allocations from d_ws
  char* base = (char*)d_ws;
  size_t off = 0;
  auto alloc = [&](size_t bytes) -> void* {
    off = (off + 255) & ~(size_t)255;
    void* p = base + off;
    off += bytes;
    return p;
  };
  float* outd = (float*)alloc((size_t)N * 4);          // out-degree -> out_n
  float* ind  = (float*)alloc((size_t)N * 4);          // in-degree  -> in_n
  float* agg1 = (float*)alloc((size_t)N * 64 * 4);     // conv1 agg; reused as t2
  float* h1   = (float*)alloc((size_t)N * 128 * 4);    // layer-1 activated
  float* agg2 = (float*)alloc((size_t)N * 64 * 4);     // conv2 agg -> h2 in-place
  float* sum1 = (float*)alloc(128 * 4);
  float* sq1  = (float*)alloc(128 * 4);
  float* sum2 = (float*)alloc(64 * 4);
  float* sq2  = (float*)alloc(64 * 4);
  float* a1   = (float*)alloc(128 * 4);
  float* b1   = (float*)alloc(128 * 4);
  float* a2   = (float*)alloc(64 * 4);
  float* b2   = (float*)alloc(64 * 4);
  float* gsum = (float*)alloc((size_t)B * 64 * 4);
  float* gcnt = (float*)alloc((size_t)B * 4);
  __bf16* W1T = (__bf16*)alloc((size_t)64 * 128 * 2);
  __bf16* W2T = (__bf16*)alloc((size_t)128 * 64 * 2);

  // zero accumulators (graph-capturable memset nodes)
  hipMemsetAsync(outd, 0, (size_t)N * 4, stream);
  hipMemsetAsync(ind, 0, (size_t)N * 4, stream);
  hipMemsetAsync(agg1, 0, (size_t)N * 64 * 4, stream);
  hipMemsetAsync(agg2, 0, (size_t)N * 64 * 4, stream);
  hipMemsetAsync(sum1, 0, (128 + 128 + 64 + 64) * 4, stream);  // sum1..sq2 contiguous
  hipMemsetAsync(gsum, 0, ((size_t)B * 64 + B) * 4, stream);   // gsum..gcnt contiguous

  // degrees + symmetric normalizers
  deg_kernel<<<(E + 255) / 256, 256, 0, stream>>>(src, dst, outd, ind, E);
  invnorm_kernel<<<(N + 255) / 256, 256, 0, stream>>>(outd, ind, N);

  // bf16 transposed weights
  cvtT_kernel<<<(64 * 128 + 255) / 256, 256, 0, stream>>>(W1, W1T, 64, 128);
  cvtT_kernel<<<(128 * 64 + 255) / 256, 256, 0, stream>>>(W2, W2T, 128, 64);

  // conv1 scatter: agg1 += node_feats[src] * (ew * out_n[src])
  scatter_kernel<<<(E * 16 + 255) / 256, 256, 0, stream>>>(node_feats, src, dst, ew, outd,
                                                           agg1, E);
  // GEMM1 + leaky
  gemm1_kernel<<<N / 16, dim3(32, 8), 0, stream>>>(agg1, ind, W1T, h1);

  // graphnorm1 coefficients
  colstats_kernel<128><<<512, 256, 0, stream>>>(h1, sum1, sq1, N);
  coeffs_kernel<<<1, 128, 0, stream>>>(sum1, sq1, gn1_gamma, gn1_beta, gn1_alpha, a1, b1,
                                       128, 1.f / (float)N);

  // GEMM2 with fused graphnorm1 affine + out_n scale; writes t2 into agg1 buffer
  float* t2 = agg1;
  gemm2_kernel<<<N / 16, dim3(32, 4), 0, stream>>>(h1, outd, a1, b1, W2T, t2);

  // conv2 scatter over 64 features: agg2 += t2[src] * ew
  scatter_kernel<<<(E * 16 + 255) / 256, 256, 0, stream>>>(t2, src, dst, ew, nullptr, agg2,
                                                           E);
  // h2 = leaky(agg2 * in_n) in-place
  finish2_kernel<<<(N * 64 + 255) / 256, 256, 0, stream>>>(agg2, ind, N * 64);

  // graphnorm2 coefficients
  colstats_kernel<64><<<512, 256, 0, stream>>>(agg2, sum2, sq2, N);
  coeffs_kernel<<<1, 64, 0, stream>>>(sum2, sq2, gn2_gamma, gn2_beta, gn2_alpha, a2, b2, 64,
                                      1.f / (float)N);

  // segment mean (affine fused)
  segment_kernel<<<(N * 16 + 255) / 256, 256, 0, stream>>>(agg2, gid, a2, b2, gsum, gcnt, N);

  // readout MLP
  readout_kernel<<<B, 64, 0, stream>>>(gsum, gcnt, L1, L2, L3, Wc, out);
}
